// Attention_59639915872858
// MI455X (gfx1250) — matrix operand order repaired and checked by reading, not compiled, not run on hardware
//
#include <hip/hip_runtime.h>
#include <hip/hip_bf16.h>
#include <cstdint>

// ---------------------------------------------------------------------------
// Types for WMMA / TDM
// ---------------------------------------------------------------------------
typedef __attribute__((ext_vector_type(16))) _Float16 v16h;
typedef __attribute__((ext_vector_type(8)))  float    v8f;
typedef __attribute__((ext_vector_type(4)))  unsigned int v4u;
typedef __attribute__((ext_vector_type(8)))  int      v8i;
typedef __attribute__((ext_vector_type(4)))  int      v4i;

union V16H { v4u u[2]; v16h h; };

#define C_ 256
#define B_ 2
#define N_ 4096
#define EPS_ 1e-5f

#if __has_builtin(__builtin_amdgcn_tensor_load_to_lds) && __has_builtin(__builtin_amdgcn_s_wait_tensorcnt)
#define USE_TDM 1
#else
#define USE_TDM 0
#endif

#if USE_TDM
// ---------------------------------------------------------------------------
// Issue one 2-D TDM tile load (f16 elements) global -> LDS.
//   tensor_d0/tile_d0 : contiguous dim (elements), stride0 in elements
//   LDS receives tile rows packed contiguously (tile_d0 elems per row).
// D# bitfields per CDNA5 ISA ch.8 (group0: count/lds_addr/global_addr/type,
// group1: data_size, tensor dims, tile dims, stride).
// ---------------------------------------------------------------------------
__device__ __forceinline__ void tdm_load_2d_f16(unsigned lds_off, const void* gaddr,
                                                unsigned tensor_d0, unsigned tensor_d1,
                                                unsigned tile_d0, unsigned tile_d1,
                                                unsigned stride0_elems) {
    unsigned long long ga = (unsigned long long)reinterpret_cast<uintptr_t>(gaddr);
    v4u g0;
    g0.x = 1u;                                           // count=1, user mode
    g0.y = lds_off;                                      // lds_addr (bytes)
    g0.z = (unsigned)(ga & 0xFFFFFFFFull);               // global_addr[31:0]
    g0.w = (unsigned)((ga >> 32) & 0x01FFFFFFull)        // global_addr[56:32]
         | (2u << 30);                                   // type = 2 ("image")
    v8i g1;
    g1[0] = (int)0x00010000u;                            // wg_mask=0, data_size=1 (2B)
    g1[1] = (int)((tensor_d0 & 0xFFFFu) << 16);          // tensor_dim0[15:0] @ bits 63:48
    g1[2] = (int)(((tensor_d0 >> 16) & 0xFFFFu)          // tensor_dim0[31:16]
         |        ((tensor_d1 & 0xFFFFu) << 16));        // tensor_dim1[15:0]
    g1[3] = (int)(((tensor_d1 >> 16) & 0xFFFFu)          // tensor_dim1[31:16]
         |        ((tile_d0 & 0xFFFFu) << 16));          // tile_dim0
    g1[4] = (int)(tile_d1 & 0xFFFFu);                    // tile_dim1 (tile_dim2=0)
    g1[5] = (int)stride0_elems;                          // tensor_dim0_stride[31:0]
    g1[6] = 0;                                           // stride hi / dim1_stride lo
    g1[7] = 0;
    v4i gz = {0, 0, 0, 0};
#if __clang_major__ >= 23
    v8i gz8 = {0, 0, 0, 0, 0, 0, 0, 0};
    __builtin_amdgcn_tensor_load_to_lds(g0, g1, gz, gz, gz8, 0);
#else
    __builtin_amdgcn_tensor_load_to_lds(g0, g1, gz, gz, 0);
#endif
}
#endif // USE_TDM

// Keys bicubic (a = -0.5) weights for frac f in [0,1)
__device__ __forceinline__ void cubw(float f, float& w0, float& w1, float& w2, float& w3) {
    float f2 = f * f;
    w1 = (1.5f * f - 2.5f) * f2 + 1.0f;                 // dist = f
    float g = 1.0f - f;
    w2 = (1.5f * g - 2.5f) * g * g + 1.0f;              // dist = 1-f
    float t0 = 1.0f + f;                                 // dist in (1,2]
    w0 = ((-0.5f * t0 + 2.5f) * t0 - 4.0f) * t0 + 2.0f;
    float t3 = 2.0f - f;
    w3 = ((-0.5f * t3 + 2.5f) * t3 - 4.0f) * t3 + 2.0f;
}
__device__ __forceinline__ int clamp63(int v) { return v < 0 ? 0 : (v > 63 ? 63 : v); }

// ---------------------------------------------------------------------------
// q projection: q[b,h,n,d] = x[b,:,n] . q_w[h*32+d,:] + q_b   (h < 4 only)
// stored f16 row-major [ (b*4+h)*N + n ][ d ]  (64B rows = WMMA A rows)
// ---------------------------------------------------------------------------
__global__ void qproj_kernel(const float* __restrict__ x,
                             const float* __restrict__ qw,
                             const float* __restrict__ qb,
                             _Float16* __restrict__ qf) {
    int idx = blockIdx.x * blockDim.x + threadIdx.x;   // [0, B*4*32*4096)
    int n = idx & 4095;
    int t = idx >> 12;
    int d = t & 31; t >>= 5;
    int h = t & 3;
    int b = t >> 2;
    int oc = h * 32 + d;
    float acc = qb[oc];
    const float* xb = x + (size_t)b * C_ * N_ + n;
    const float* w  = qw + (size_t)oc * C_;
    #pragma unroll 8
    for (int c = 0; c < C_; ++c) acc += w[c] * xb[(size_t)c * N_];
    qf[(((size_t)b * 4 + h) * N_ + n) * 32 + d] = (_Float16)acc;
}

// ---------------------------------------------------------------------------
// z = BN2( pw[c] * relu(BN1( dwconv_stride(x) )) )
// ---------------------------------------------------------------------------
template <int KS, int STRIDE, int PAD, int OHW>
__global__ void dwz_kernel(const float* __restrict__ x,
                           const float* __restrict__ dw_w,
                           const float* __restrict__ bn1,
                           const float* __restrict__ pw_w,
                           const float* __restrict__ bn2,
                           float* __restrict__ z) {
    int idx = blockIdx.x * blockDim.x + threadIdx.x;   // [0, B*C*OHW*OHW)
    int ox = idx % OHW;
    int t  = idx / OHW;
    int oy = t % OHW; t /= OHW;
    int c  = t % C_;
    int b  = t / C_;
    const float* xb = x + ((size_t)b * C_ + c) * 64 * 64;
    const float* wt = dw_w + (size_t)c * KS * KS;
    float acc = 0.f;
    #pragma unroll
    for (int ky = 0; ky < KS; ++ky) {
        int iy = oy * STRIDE - PAD + ky;
        if (iy < 0 || iy >= 64) continue;
        #pragma unroll
        for (int kx = 0; kx < KS; ++kx) {
            int ix = ox * STRIDE - PAD + kx;
            if (ix < 0 || ix >= 64) continue;
            acc += xb[iy * 64 + ix] * wt[ky * KS + kx];
        }
    }
    float a = (acc - bn1[2 * C_ + c]) * bn1[c] * rsqrtf(bn1[3 * C_ + c] + EPS_) + bn1[C_ + c];
    a = fmaxf(a, 0.f);
    a *= pw_w[c];
    a = (a - bn2[2 * C_ + c]) * bn2[c] * rsqrtf(bn2[3 * C_ + c] + EPS_) + bn2[C_ + c];
    z[idx] = a;
}

// ---------------------------------------------------------------------------
// y = dw3x3(z) + local_b + z
// ---------------------------------------------------------------------------
template <int OHW>
__global__ void local_kernel(const float* __restrict__ z,
                             const float* __restrict__ lw,
                             const float* __restrict__ lb,
                             float* __restrict__ y) {
    int idx = blockIdx.x * blockDim.x + threadIdx.x;
    int ox = idx % OHW;
    int t  = idx / OHW;
    int oy = t % OHW; t /= OHW;
    int c  = t % C_;
    int b  = t / C_;
    const float* zb = z + ((size_t)b * C_ + c) * OHW * OHW;
    const float* wt = lw + (size_t)c * 9;
    float acc = lb[c] + zb[oy * OHW + ox];
    #pragma unroll
    for (int ky = 0; ky < 3; ++ky) {
        int iy = oy - 1 + ky;
        if (iy < 0 || iy >= OHW) continue;
        #pragma unroll
        for (int kx = 0; kx < 3; ++kx) {
            int ix = ox - 1 + kx;
            if (ix < 0 || ix >= OHW) continue;
            acc += zb[iy * OHW + ix] * wt[ky * 3 + kx];
        }
    }
    y[idx] = acc;
}

// ---------------------------------------------------------------------------
// kv projection + scatter into WMMA-friendly f16 K / V
//   K stored [b][h][l][d] (d contiguous, 64B rows -> B-operand of QK^T)
//   V stored [b][h][d][l] (l contiguous        -> B-operand of AV)
// reshape rule: channel c' (within head) -> d = c'>>1, l = (c'&1)*HW + s
// ---------------------------------------------------------------------------
template <int HW>
__global__ void kvproj_kernel(const float* __restrict__ y,
                              const float* __restrict__ kv_w,
                              const float* __restrict__ kv_b,
                              _Float16* __restrict__ Kf,
                              _Float16* __restrict__ Vf) {
    const int L = 2 * HW;
    int idx = blockIdx.x * blockDim.x + threadIdx.x;   // [0, B*512*HW)
    int s = idx % HW;
    int t = idx / HW;
    int o = t % 512;
    int b = t / 512;
    float acc = kv_b[o];
    const float* yb = y + (size_t)b * C_ * HW + s;
    const float* w  = kv_w + (size_t)o * C_;
    #pragma unroll 8
    for (int c = 0; c < C_; ++c) acc += w[c] * yb[(size_t)c * HW];
    if (o < 256) {
        int h = o >> 6, cp = o & 63, d = cp >> 1;
        int l = (cp & 1) ? (HW + s) : s;
        Kf[(((size_t)b * 4 + h) * L + l) * 32 + d] = (_Float16)acc;
    } else {
        int op = o - 256;
        int h = op >> 6, cp = op & 63, d = cp >> 1;
        int l = (cp & 1) ? (HW + s) : s;
        Vf[(((size_t)b * 4 + h) * 32 + d) * (size_t)L + l] = (_Float16)acc;
    }
}

// ---------------------------------------------------------------------------
// Flash attention: WMMA f32_16x16x32_f16, online softmax, bicubic RPE.
// Block = 128 threads (4 wave32). Each wave owns 32 query rows (two A tiles),
// so every staged K/V tile feeds 2 WMMAs (8 WMMAs per 32-key chunk).
// K/V chunks staged to LDS by the Tensor Data Mover, double buffered so
// chunk i+1's DMA overlaps chunk i's WMMAs. grid = B * 4 * (N/128).
// ---------------------------------------------------------------------------
template <int L, int EBASE, int SLICE>
__global__ void __launch_bounds__(128)
attn_kernel(const _Float16* __restrict__ qf,
            const _Float16* __restrict__ Kf,
            const _Float16* __restrict__ Vf,
            const float* __restrict__ rpe,
            float* __restrict__ out) {
    constexpr float SCL = 0.17677669529663687f;   // 32^-0.5
    __shared__ float    tmp[4][32 * 64];          // per-wave row-interpolated RPE (32 KB)
    __shared__ _Float16 attnX[4][32 * 32];        // per-wave P tile exchange      (8 KB)
#if USE_TDM
    __shared__ _Float16 Kbuf[2][32 * 32];         // double-buffered K chunk       (4 KB)
    __shared__ _Float16 Vbuf[2][32 * 32];         // double-buffered V chunk       (4 KB)
#endif

    const int wave = threadIdx.x >> 5;
    const int lane = threadIdx.x & 31;
    const int half = lane >> 4;
    const int c15  = lane & 15;

    int blk = blockIdx.x;                 // ((b*4 + h)*32 + nb)
    const int nb = blk & 31; blk >>= 5;
    const int h  = blk & 3;
    const int b  = blk >> 2;
    const int n0 = nb * 128 + wave * 32;  // 32 query rows per wave
    const size_t bh = (size_t)b * 4 + h;

    const _Float16* Kbh = Kf + bh * (size_t)L * 32;
    const _Float16* Vbh = Vf + bh * 32 * (size_t)L;

    // ---- load Q rows as two WMMA A operands (rows n0..n0+15, n0+16..n0+31)
    V16H aqU0, aqU1;
    {
        const _Float16* qrow = qf + (bh * N_ + n0 + c15) * 32;
        aqU0.u[0] = *(const v4u*)((const char*)qrow + half * 32);
        aqU0.u[1] = *(const v4u*)((const char*)qrow + 16 + half * 32);
    }
    {
        const _Float16* qrow = qf + (bh * N_ + n0 + 16 + c15) * 32;
        aqU1.u[0] = *(const v4u*)((const char*)qrow + half * 32);
        aqU1.u[1] = *(const v4u*)((const char*)qrow + 16 + half * 32);
    }
    const v16h aq0 = aqU0.h;
    const v16h aq1 = aqU1.h;

    // ---- separable bicubic: interpolate along N once per wave into LDS
    const float* rsrc = rpe + ((size_t)b * 2 + SLICE) * 4096;
    for (int i = lane; i < 2048; i += 32) {
        int r = i >> 6, cc = i & 63;                   // r in [0,32)
        float u  = ((float)(n0 + r) + 0.5f) * (1.0f / 64.0f) - 0.5f;
        float fu = floorf(u);
        int   iy = (int)fu;
        float f  = u - fu;
        float w0, w1, w2, w3; cubw(f, w0, w1, w2, w3);
        float acc = w0 * rsrc[clamp63(iy - 1) * 64 + cc]
                  + w1 * rsrc[clamp63(iy    ) * 64 + cc]
                  + w2 * rsrc[clamp63(iy + 1) * 64 + cc]
                  + w3 * rsrc[clamp63(iy + 2) * 64 + cc];
        tmp[wave][r * 64 + cc] = acc;
    }

#if USE_TDM
    // prologue: DMA chunk 0 into buffer 0 (K tile: rows=l, 32x32; V tile: rows=d)
    if (wave == 0) {
        tdm_load_2d_f16((unsigned)reinterpret_cast<uintptr_t>((void*)&Kbuf[0][0]),
                        (const void*)Kbh, 32u, (unsigned)L, 32u, 32u, 32u);
        tdm_load_2d_f16((unsigned)reinterpret_cast<uintptr_t>((void*)&Vbuf[0][0]),
                        (const void*)Vbh, (unsigned)L, 32u, 32u, 32u, (unsigned)L);
    }
#endif
    __syncthreads();

    float M[2][8], S[2][8];
    v8f oacc[2][2] = {{{}, {}}, {{}, {}}};   // [row group][d tile]
    const v8f vzero = {};
    #pragma unroll
    for (int g = 0; g < 2; ++g)
        #pragma unroll
        for (int r = 0; r < 8; ++r) { M[g][r] = -1e30f; S[g][r] = 0.f; }

    const float* trow = &tmp[wave][0];

    for (int l0 = 0; l0 < L; l0 += 32) {
#if USE_TDM
        const int bi = (l0 >> 5) & 1;
        // all consumers of buffer bi^1 (previous chunk) passed the last barrier;
        // wave 0 launches the next chunk's DMA into it, then waits until the
        // CURRENT chunk (2 in-order TDM ops back) has landed.
        if (wave == 0) {
            if (l0 + 32 < L) {
                tdm_load_2d_f16((unsigned)reinterpret_cast<uintptr_t>((void*)&Kbuf[bi ^ 1][0]),
                                (const void*)(Kbh + (size_t)(l0 + 32) * 32),
                                32u, (unsigned)(L - l0 - 32), 32u, 32u, 32u);
                tdm_load_2d_f16((unsigned)reinterpret_cast<uintptr_t>((void*)&Vbuf[bi ^ 1][0]),
                                (const void*)(Vbh + (size_t)(l0 + 32)),
                                (unsigned)(L - l0 - 32), 32u, 32u, 32u, (unsigned)L);
                __builtin_amdgcn_s_wait_tensorcnt(2);
            } else {
                __builtin_amdgcn_s_wait_tensorcnt(0);
            }
        }
        __syncthreads();   // publish chunk l0 to all waves
        const _Float16* Kl = &Kbuf[bi][0];   // [l][d] rows, 64B each
        const _Float16* Vl = &Vbuf[bi][0];   // [d][l] rows, 64B each
        V16H kb0, kb1;
        {
            const char* p = (const char*)(Kl + (size_t)c15 * 32) + half * 32;
            kb0.u[0] = *(const v4u*)p; kb0.u[1] = *(const v4u*)(p + 16);
        }
        {
            const char* p = (const char*)(Kl + (size_t)(16 + c15) * 32) + half * 32;
            kb1.u[0] = *(const v4u*)p; kb1.u[1] = *(const v4u*)(p + 16);
        }
#else
        if (l0 + 32 < L) {
            __builtin_prefetch((const void*)(Kbh + (size_t)(l0 + 32) * 32), 0, 1);
            __builtin_prefetch((const void*)(Vbh + (size_t)c15 * L + l0 + 32), 0, 1);
        }
        V16H kb0, kb1;
        {
            const char* p = (const char*)(Kbh + (size_t)(l0 + c15) * 32) + half * 32;
            kb0.u[0] = *(const v4u*)p; kb0.u[1] = *(const v4u*)(p + 16);
        }
        {
            const char* p = (const char*)(Kbh + (size_t)(l0 + 16 + c15) * 32) + half * 32;
            kb1.u[0] = *(const v4u*)p; kb1.u[1] = *(const v4u*)(p + 16);
        }
#endif
        // ---- 4 QK^T WMMAs: both A tiles reuse both K tiles
        v8f sQK[2][2];
        sQK[0][0] = __builtin_amdgcn_wmma_f32_16x16x32_f16(false, aq0, false, kb0.h, (short)0, vzero, false, false);
        sQK[0][1] = __builtin_amdgcn_wmma_f32_16x16x32_f16(false, aq0, false, kb1.h, (short)0, vzero, false, false);
        sQK[1][0] = __builtin_amdgcn_wmma_f32_16x16x32_f16(false, aq1, false, kb0.h, (short)0, vzero, false, false);
        sQK[1][1] = __builtin_amdgcn_wmma_f32_16x16x32_f16(false, aq1, false, kb1.h, (short)0, vzero, false, false);

        // ---- scale + RPE (column bicubic), track chunk row max per group
        float cm[2][8];
        #pragma unroll
        for (int g = 0; g < 2; ++g)
            #pragma unroll
            for (int r = 0; r < 8; ++r) cm[g][r] = -1e30f;

        #pragma unroll
        for (int t = 0; t < 2; ++t) {
            int   l  = l0 + t * 16 + c15;
            float u  = ((float)l + 0.5f) * (64.0f / (float)L) - 0.5f;
            float fu = floorf(u); int ix = (int)fu; float f = u - fu;
            float w0, w1, w2, w3; cubw(f, w0, w1, w2, w3);
            int c0 = clamp63(ix - 1), c1 = clamp63(ix), c2 = clamp63(ix + 1), c3 = clamp63(ix + 2);
            #pragma unroll
            for (int g = 0; g < 2; ++g) {
                #pragma unroll
                for (int r = 0; r < 8; ++r) {
                    const float* tr = trow + (g * 16 + r + 8 * half) * 64;
                    float rv = w0 * tr[c0] + w1 * tr[c1] + w2 * tr[c2] + w3 * tr[c3];
                    float vv = sQK[g][t][r] * SCL + rv;
                    sQK[g][t][r] = vv;
                    cm[g][r] = fmaxf(cm[g][r], vv);
                }
            }
        }
        // row-max reduction within each 16-lane half
        #pragma unroll
        for (int g = 0; g < 2; ++g) {
            #pragma unroll
            for (int r = 0; r < 8; ++r) {
                float v = cm[g][r];
                v = fmaxf(v, __shfl_xor(v, 1));
                v = fmaxf(v, __shfl_xor(v, 2));
                v = fmaxf(v, __shfl_xor(v, 4));
                v = fmaxf(v, __shfl_xor(v, 8));
                cm[g][r] = v;
            }
        }
        // online softmax update
        float ps[2][8];
        #pragma unroll
        for (int g = 0; g < 2; ++g) {
            #pragma unroll
            for (int r = 0; r < 8; ++r) {
                float nm    = fmaxf(M[g][r], cm[g][r]);
                float alpha = __expf(M[g][r] - nm);
                M[g][r] = nm;
                S[g][r] *= alpha;
                oacc[g][0][r] *= alpha;
                oacc[g][1][r] *= alpha;
                float p0 = __expf(sQK[g][0][r] - nm);
                float p1 = __expf(sQK[g][1][r] - nm);
                sQK[g][0][r] = p0; sQK[g][1][r] = p1;
                ps[g][r] = p0 + p1;
            }
            #pragma unroll
            for (int r = 0; r < 8; ++r) {
                float v = ps[g][r];
                v += __shfl_xor(v, 1);
                v += __shfl_xor(v, 2);
                v += __shfl_xor(v, 4);
                v += __shfl_xor(v, 8);
                S[g][r] += v;
            }
        }
        // ---- P (C layout) -> LDS -> two A operands
        _Float16* ax = &attnX[wave][0];
        #pragma unroll
        for (int g = 0; g < 2; ++g)
            #pragma unroll
            for (int r = 0; r < 8; ++r) {
                ax[(g * 16 + r + 8 * half) * 32 + c15]      = (_Float16)sQK[g][0][r];
                ax[(g * 16 + r + 8 * half) * 32 + 16 + c15] = (_Float16)sQK[g][1][r];
            }
        __syncthreads();
        V16H pa0, pa1;
        {
            const char* p = (const char*)(ax + (size_t)c15 * 32);
            pa0.u[0] = *(const v4u*)(p + half * 32);
            pa0.u[1] = *(const v4u*)(p + 16 + half * 32);
        }
        {
            const char* p = (const char*)(ax + (size_t)(16 + c15) * 32);
            pa1.u[0] = *(const v4u*)(p + half * 32);
            pa1.u[1] = *(const v4u*)(p + 16 + half * 32);
        }
        // ---- V tiles as B operands, 4 AV WMMAs accumulate O
        V16H vb0, vb1;
#if USE_TDM
        {
            const _Float16* p = Vl + (size_t)c15 * 32 + half * 16;
            vb0.u[0] = *(const v4u*)p; vb0.u[1] = *(const v4u*)(p + 8);
        }
        {
            const _Float16* p = Vl + (size_t)(16 + c15) * 32 + half * 16;
            vb1.u[0] = *(const v4u*)p; vb1.u[1] = *(const v4u*)(p + 8);
        }
#else
        {
            const _Float16* p = Vbh + (size_t)c15 * L + l0 + half * 16;
            vb0.u[0] = *(const v4u*)p; vb0.u[1] = *(const v4u*)(p + 8);
        }
        {
            const _Float16* p = Vbh + (size_t)(16 + c15) * L + l0 + half * 16;
            vb1.u[0] = *(const v4u*)p; vb1.u[1] = *(const v4u*)(p + 8);
        }
#endif
        oacc[0][0] = __builtin_amdgcn_wmma_f32_16x16x32_f16(false, pa0.h, false, vb0.h, (short)0, oacc[0][0], false, false);
        oacc[0][1] = __builtin_amdgcn_wmma_f32_16x16x32_f16(false, pa0.h, false, vb1.h, (short)0, oacc[0][1], false, false);
        oacc[1][0] = __builtin_amdgcn_wmma_f32_16x16x32_f16(false, pa1.h, false, vb0.h, (short)0, oacc[1][0], false, false);
        oacc[1][1] = __builtin_amdgcn_wmma_f32_16x16x32_f16(false, pa1.h, false, vb1.h, (short)0, oacc[1][1], false, false);
        __syncthreads();   // all waves done with LDS buffers before re-fill/re-use
    }

    // ---- epilogue: normalize and scatter into concat layout
    float* ob = out + (size_t)b * (C_ * N_) + (size_t)h * (64 * N_);
    #pragma unroll
    for (int g = 0; g < 2; ++g)
        #pragma unroll
        for (int r = 0; r < 8; ++r) {
            int   n   = n0 + g * 16 + r + 8 * half;
            float inv = 1.0f / S[g][r];
            ob[(size_t)n * 64 + EBASE + c15]      = oacc[g][0][r] * inv;
            ob[(size_t)n * 64 + EBASE + 16 + c15] = oacc[g][1][r] * inv;
        }
}

// ---------------------------------------------------------------------------
// Host launcher
// ---------------------------------------------------------------------------
extern "C" void kernel_launch(void* const* d_in, const int* in_sizes, int n_in,
                              void* d_out, int out_size, void* d_ws, size_t ws_size,
                              hipStream_t stream) {
    (void)in_sizes; (void)n_in; (void)out_size; (void)ws_size;
    const float* x        = (const float*)d_in[0];
    const float* rpe      = (const float*)d_in[1];
    const float* q_w      = (const float*)d_in[2];
    const float* q_b      = (const float*)d_in[3];
    const float* kv_w     = (const float*)d_in[4];
    const float* kv_b     = (const float*)d_in[5];
    const float* sr1_dw_w = (const float*)d_in[6];
    const float* sr1_bn1  = (const float*)d_in[7];
    const float* sr1_pw_w = (const float*)d_in[8];
    const float* sr1_bn2  = (const float*)d_in[9];
    const float* sr2_dw_w = (const float*)d_in[10];
    const float* sr2_bn1  = (const float*)d_in[11];
    const float* sr2_pw_w = (const float*)d_in[12];
    const float* sr2_bn2  = (const float*)d_in[13];
    const float* local_w  = (const float*)d_in[14];
    const float* local_b  = (const float*)d_in[15];
    float* out = (float*)d_out;

    char* ws = (char*)d_ws;
    _Float16* qF = (_Float16*)(ws);                    // 2 MB
    float*    z1 = (float*)(ws + 2097152);             // 512 KB
    float*    y1 = (float*)(ws + 2621440);             // 512 KB
    float*    z2 = (float*)(ws + 3145728);             // 2 MB
    float*    y2 = (float*)(ws + 5242880);             // 2 MB
    _Float16* K1 = (_Float16*)(ws + 7340032);          // 256 KB
    _Float16* V1 = (_Float16*)(ws + 7602176);          // 256 KB
    _Float16* K2 = (_Float16*)(ws + 7864320);          // 1 MB
    _Float16* V2 = (_Float16*)(ws + 8912896);          // 1 MB

    // q projection (heads 0..3): B*4*32*4096 threads
    qproj_kernel<<<4096, 256, 0, stream>>>(x, q_w, q_b, qF);

    // ---- branch 1: 7x7 stride 4 pad 3 -> 16x16, L = 512
    dwz_kernel<7, 4, 3, 16><<<512, 256, 0, stream>>>(x, sr1_dw_w, sr1_bn1, sr1_pw_w, sr1_bn2, z1);
    local_kernel<16><<<512, 256, 0, stream>>>(z1, local_w, local_b, y1);
    kvproj_kernel<256><<<1024, 256, 0, stream>>>(y1, kv_w, kv_b, K1, V1);
    attn_kernel<512, 0, 0><<<256, 128, 0, stream>>>(qF, K1, V1, rpe, out);

    // ---- branch 2: 5x5 stride 2 pad 2 -> 32x32, L = 2048
    dwz_kernel<5, 2, 2, 32><<<2048, 256, 0, stream>>>(x, sr2_dw_w, sr2_bn1, sr2_pw_w, sr2_bn2, z2);
    local_kernel<32><<<2048, 256, 0, stream>>>(z2, local_w, local_b, y2);
    kvproj_kernel<1024><<<4096, 256, 0, stream>>>(y2, kv_w, kv_b, K2, V2);
    attn_kernel<2048, 32, 1><<<256, 128, 0, stream>>>(qF, K2, V2, rpe, out);
}
